// PointNetXS_35708358099260
// MI455X (gfx1250) — matrix operand order repaired and checked by reading, not compiled, not run on hardware
//
#include <hip/hip_runtime.h>
#include <hip/hip_bf16.h>
#include <math.h>

// ---------------------------------------------------------------------------
// PointNet-XS head for MI455X (gfx1250, wave32, WMMA).
// GEMM2 (N x 256 x 64) runs on v_wmma_f32_16x16x32_f16 (f16 in, f32 acc).
// BN1 stats computed analytically from the 3x3 second-moment of feats.
// BN2 stats computed in a first WMMA pass; y2 recomputed in the second pass
// (cheaper than 128MB of y2 spill traffic at 23.3 TB/s HBM).
// W2/W3 WMMA fragments are staged in LDS; kT loops stay rolled so the live
// register set per iteration is ~80 VGPRs (no scratch spills).
// ---------------------------------------------------------------------------

typedef __attribute__((ext_vector_type(16))) _Float16 v16h;
typedef __attribute__((ext_vector_type(8)))  float    v8f;

union FragA { v16h v; _Float16 e[16]; };
union FragB { v16h v; uint4 q[2]; };

// workspace layout (dword offsets)
#define STATS_OFF 0      // 9 floats: sum x,y,z ; moments xx,xy,xz,yy,yz,zz
#define SEG_OFF   16     // 16*128 ordered-uint segment max
#define A1_OFF    2064   // 128 x float4 : BN1-folded affine map  x1 = A1.xyz . f + A1.w
#define SUM_OFF   2576   // 64 floats : sum of y2 per channel
#define SQ_OFF    2640   // 64 floats : sum of y2^2 per channel
#define W2F_OFF   2704   // 8192 dwords : W2 as f16 WMMA B-fragments
#define W3F_OFF   10896  // 512 dwords  : W3 as f16 WMMA B-fragments
#define WS_DWORDS 11408
#define NB 16            // num_batches (fixed by setup_inputs)
#define EPS 1e-5f

// order-preserving float<->uint mapping for atomicMax on floats
__device__ __forceinline__ unsigned enc(float f) {
  unsigned u = __float_as_uint(f);
  return (u & 0x80000000u) ? ~u : (u | 0x80000000u);
}
__device__ __forceinline__ float dec(unsigned u) {
  return (u & 0x80000000u) ? __uint_as_float(u ^ 0x80000000u)
                           : __uint_as_float(~u);
}
__device__ __forceinline__ unsigned pack2h(float a, float b) {
  union { _Float16 h[2]; unsigned u; } x;
  x.h[0] = (_Float16)a; x.h[1] = (_Float16)b;
  return x.u;
}

// -------------------------- kernel 1: init scratch --------------------------
__global__ void k_init(unsigned* wsU) {
  for (int i = threadIdx.x; i < W2F_OFF; i += blockDim.x) wsU[i] = 0u;
}

// ------------------- kernel 2: feats mean + second moment -------------------
__global__ void k_fstats(const float* __restrict__ feats, float* wsF, int n) {
  __shared__ float red[9];
  int tid = threadIdx.x;
  if (tid < 9) red[tid] = 0.f;
  __syncthreads();
  float s0=0,s1=0,s2=0,m00=0,m01=0,m02=0,m11=0,m12=0,m22=0;
  #pragma unroll 1
  for (int p = blockIdx.x * blockDim.x + tid; p < n; p += gridDim.x * blockDim.x) {
    float x = feats[3*p], y = feats[3*p+1], z = feats[3*p+2];
    s0+=x; s1+=y; s2+=z;
    m00+=x*x; m01+=x*y; m02+=x*z; m11+=y*y; m12+=y*z; m22+=z*z;
  }
  atomicAdd(&red[0],s0);  atomicAdd(&red[1],s1);  atomicAdd(&red[2],s2);
  atomicAdd(&red[3],m00); atomicAdd(&red[4],m01); atomicAdd(&red[5],m02);
  atomicAdd(&red[6],m11); atomicAdd(&red[7],m12); atomicAdd(&red[8],m22);
  __syncthreads();
  if (tid < 9) atomicAdd(&wsF[STATS_OFF + tid], red[tid]);
}

// --- kernel 3: BN1-folded affine map + repack W2/W3 into WMMA fragments -----
// B-fragment layout for v_wmma_f32_16x16x32_f16 (wave32):
//   lane (0..15 | 16..31) holds column N = lane&15 ;
//   vgpr v holds K pair (k, k+1), k = 2(v&3) + (v>=4?16:0) for lanes 0-15, k+8 for 16-31.
__global__ void k_prep(const float* __restrict__ W1, const float* __restrict__ g1,
                       const float* __restrict__ b1, const float* __restrict__ W2,
                       const float* __restrict__ W3, float* wsF, unsigned* wsU, int n) {
  int tid = threadIdx.x;
  float inv = 1.0f / (float)n;
  // BN1 params: mean_c = w.mu ; var_c = w^T M w - mean_c^2   (exact, biased)
  for (int c = tid; c < 128; c += blockDim.x) {
    float mu0 = wsF[0]*inv, mu1 = wsF[1]*inv, mu2 = wsF[2]*inv;
    float M00 = wsF[3]*inv, M01 = wsF[4]*inv, M02 = wsF[5]*inv;
    float M11 = wsF[6]*inv, M12 = wsF[7]*inv, M22 = wsF[8]*inv;
    float w0 = W1[c*3+0], w1 = W1[c*3+1], w2 = W1[c*3+2];
    float mean = w0*mu0 + w1*mu1 + w2*mu2;
    float ey2  = w0*w0*M00 + w1*w1*M11 + w2*w2*M22
               + 2.f*(w0*w1*M01 + w0*w2*M02 + w1*w2*M12);
    float var  = ey2 - mean*mean;
    float sc   = g1[c] * rsqrtf(var + EPS);
    float sh   = b1[c] - mean * sc;
    float4* a1 = (float4*)(wsF + A1_OFF);
    a1[c] = make_float4(sc*w0, sc*w1, sc*w2, sh);
  }
  // W2 fragments: index = ((nT*8+kT)*32 + lane)*8 + v
  for (int t = tid; t < 8192; t += blockDim.x) {
    int v = t & 7, lane = (t >> 3) & 31, kT = (t >> 8) & 7, nT = t >> 11;
    int hi = lane >> 4;
    int nn = nT*16 + (lane & 15);
    int k  = kT*32 + 2*(v & 3) + ((v >= 4) ? 16 : 0) + (hi ? 8 : 0);
    wsU[W2F_OFF + t] = pack2h(W2[nn*256 + k], W2[nn*256 + k + 1]);
  }
  // W3 fragments: index = (kT*32 + lane)*8 + v  (single N-tile of 16)
  for (int t = tid; t < 512; t += blockDim.x) {
    int v = t & 7, lane = (t >> 3) & 31, kT = t >> 8;
    int hi = lane >> 4;
    int j  = lane & 15;
    int k  = kT*32 + 2*(v & 3) + ((v >= 4) ? 16 : 0) + (hi ? 8 : 0);
    wsU[W3F_OFF + t] = pack2h(W3[j*64 + k], W3[j*64 + k + 1]);
  }
}

// ----------------- kernel 4: per-batch segment max of x1 --------------------
__global__ void k_segmax(const float* __restrict__ feats, const int* __restrict__ bidx,
                         const float* wsF, unsigned* wsU, int n) {
  __shared__ unsigned lmax[NB * 128];
  __shared__ float4 a1s[128];
  int tid = threadIdx.x;
  for (int i = tid; i < NB*128; i += blockDim.x) lmax[i] = 0u;  // enc < any finite
  for (int i = tid; i < 128; i += blockDim.x)
    a1s[i] = ((const float4*)(wsF + A1_OFF))[i];
  __syncthreads();
  int p = blockIdx.x * blockDim.x + tid;
  if (p < n) {
    float f0 = feats[3*p], f1 = feats[3*p+1], f2 = feats[3*p+2];
    int b = bidx[p] & (NB-1);
    #pragma unroll 4
    for (int c = 0; c < 128; ++c) {
      float4 a = a1s[c];
      float x = a.x*f0 + a.y*f1 + a.z*f2 + a.w;
      atomicMax(&lmax[b*128 + c], enc(x));
    }
  }
  __syncthreads();
  for (int i = tid; i < NB*128; i += blockDim.x)
    atomicMax(&wsU[SEG_OFF + i], lmax[i]);
}

// ---- shared device routine: one wave computes y2 for a 16-point tile -------
// A-fragment layout (16-bit A 16x32): lanes 0-15 hold row M=lane with
// K = {2(v&3)+ (v>=4?16:0)} pairs; lanes 16-31 same rows, K offset +8.
// kT loops intentionally NOT unrolled: keeps live VGPRs ~80, no spills.
__device__ __forceinline__ void tile_gemm2(
    int tile, int lane,
    const float* __restrict__ feats, const int* __restrict__ bidx,
    const float4* a1, const float* smax, const unsigned* w2s, v8f acc[4]) {
  int pl = lane & 15, hi = lane >> 4;
  int p = tile * 16 + pl;
  float f0 = feats[3*p], f1 = feats[3*p+1], f2 = feats[3*p+2];
  int b = bidx[p] & (NB-1);
  #pragma unroll
  for (int i = 0; i < 4; ++i)
    #pragma unroll
    for (int j = 0; j < 8; ++j) acc[i][j] = 0.f;
  // ---- kT 0..3 : x1 half, recomputed from BN1-folded affine map ----
  #pragma unroll 1
  for (int kT = 0; kT < 4; ++kT) {
    FragA a;
    #pragma unroll
    for (int v = 0; v < 8; ++v) {
      int k = kT*32 + 2*(v & 3) + ((v >= 4) ? 16 : 0) + (hi ? 8 : 0);
      float4 c0 = a1[k], c1 = a1[k+1];
      a.e[2*v]   = (_Float16)(c0.x*f0 + c0.y*f1 + c0.z*f2 + c0.w);
      a.e[2*v+1] = (_Float16)(c1.x*f0 + c1.y*f1 + c1.z*f2 + c1.w);
    }
    #pragma unroll
    for (int nT = 0; nT < 4; ++nT) {
      FragB bf;
      const uint4* src = (const uint4*)(w2s + ((nT*8 + kT)*32 + lane)*8);
      bf.q[0] = src[0]; bf.q[1] = src[1];
      acc[nT] = __builtin_amdgcn_wmma_f32_16x16x32_f16(
          false, a.v, false, bf.v, (short)0, acc[nT], false, false);
    }
  }
  // ---- kT 4..7 : concat half, gather broadcast seg-max ----
  #pragma unroll 1
  for (int kT = 4; kT < 8; ++kT) {
    FragA a;
    #pragma unroll
    for (int v = 0; v < 8; ++v) {
      int k = kT*32 + 2*(v & 3) + ((v >= 4) ? 16 : 0) + (hi ? 8 : 0) - 128;
      a.e[2*v]   = (_Float16)smax[b*128 + k];
      a.e[2*v+1] = (_Float16)smax[b*128 + k + 1];
    }
    #pragma unroll
    for (int nT = 0; nT < 4; ++nT) {
      FragB bf;
      const uint4* src = (const uint4*)(w2s + ((nT*8 + kT)*32 + lane)*8);
      bf.q[0] = src[0]; bf.q[1] = src[1];
      acc[nT] = __builtin_amdgcn_wmma_f32_16x16x32_f16(
          false, a.v, false, bf.v, (short)0, acc[nT], false, false);
    }
  }
}

// -------------- kernel 5: WMMA pass 1 -> BN2 sum / sumsq --------------------
__global__ void __launch_bounds__(128, 1)
k_y2stats(const float* __restrict__ feats, const int* __restrict__ bidx,
          unsigned* wsU, float* wsF, int n) {
  __shared__ float4 a1s[128];
  __shared__ float smaxs[NB * 128];
  __shared__ unsigned w2s[8192];       // 32KB: W2 WMMA fragments
  __shared__ float bsum[64], bsq[64];
  int tid = threadIdx.x;
  for (int i = tid; i < 128; i += blockDim.x)
    a1s[i] = ((const float4*)(wsF + A1_OFF))[i];
  for (int i = tid; i < NB*128; i += blockDim.x)
    smaxs[i] = dec(wsU[SEG_OFF + i]);
  for (int i = tid; i < 2048; i += blockDim.x)
    ((uint4*)w2s)[i] = ((const uint4*)(wsU + W2F_OFF))[i];
  for (int i = tid; i < 64; i += blockDim.x) { bsum[i] = 0.f; bsq[i] = 0.f; }
  __syncthreads();
  int lane = tid & 31, wv = tid >> 5;
  int gw = blockIdx.x * (blockDim.x >> 5) + wv;
  int nw = gridDim.x * (blockDim.x >> 5);
  int tiles = n / 16, pl = lane & 15;
  #pragma unroll 1
  for (int t = gw; t < tiles; t += nw) {
    v8f acc[4];
    tile_gemm2(t, lane, feats, bidx, a1s, smaxs, w2s, acc);
    #pragma unroll
    for (int nT = 0; nT < 4; ++nT) {
      float s = 0.f, q = 0.f;
      #pragma unroll
      for (int r = 0; r < 8; ++r) { float x = acc[nT][r]; s += x; q += x*x; }
      s += __shfl_xor(s, 16, 32);
      q += __shfl_xor(q, 16, 32);
      if (lane < 16) {
        atomicAdd(&bsum[nT*16 + pl], s);
        atomicAdd(&bsq [nT*16 + pl], q);
      }
    }
  }
  __syncthreads();
  for (int i = tid; i < 64; i += blockDim.x) {
    atomicAdd(wsF + SUM_OFF + i, bsum[i]);
    atomicAdd(wsF + SQ_OFF  + i, bsq[i]);
  }
}

// ---- kernel 6: WMMA pass 2: BN2 + relu + GEMM3 (via LDS transpose) ---------
__global__ void __launch_bounds__(128, 1)
k_final(const float* __restrict__ feats, const int* __restrict__ bidx,
        const float* __restrict__ g2, const float* __restrict__ b2,
        const float* __restrict__ b3, unsigned* wsU, float* wsF,
        float* __restrict__ out, int n) {
  __shared__ float4 a1s[128];
  __shared__ float smaxs[NB * 128];
  __shared__ unsigned w2s[8192];       // 32KB
  __shared__ unsigned w3s[512];        // 2KB
  __shared__ _Float16 x2s[4][16][64];  // per-wave transpose buffer, 8KB
  int tid = threadIdx.x;
  for (int i = tid; i < 128; i += blockDim.x)
    a1s[i] = ((const float4*)(wsF + A1_OFF))[i];
  for (int i = tid; i < NB*128; i += blockDim.x)
    smaxs[i] = dec(wsU[SEG_OFF + i]);
  for (int i = tid; i < 2048; i += blockDim.x)
    ((uint4*)w2s)[i] = ((const uint4*)(wsU + W2F_OFF))[i];
  for (int i = tid; i < 512; i += blockDim.x)
    w3s[i] = wsU[W3F_OFF + i];
  __syncthreads();
  int lane = tid & 31, wv = tid >> 5;
  int pl = lane & 15, hi = lane >> 4;
  float inv = 1.0f / (float)n;
  float s2[4], t2[4];
  #pragma unroll
  for (int nT = 0; nT < 4; ++nT) {
    int c = nT*16 + pl;
    float mean = wsF[SUM_OFF + c] * inv;
    float var  = wsF[SQ_OFF + c] * inv - mean*mean;
    float sc   = g2[c] * rsqrtf(var + EPS);
    s2[nT] = sc; t2[nT] = b2[c] - mean * sc;
  }
  float bias = b3[pl];
  int gw = blockIdx.x * (blockDim.x >> 5) + wv;
  int nw = gridDim.x * (blockDim.x >> 5);
  int tiles = n / 16;
  int iters = (tiles + nw - 1) / nw;       // uniform trip count for barriers
  #pragma unroll 1
  for (int it = 0; it < iters; ++it) {
    int t = gw + it * nw;
    bool valid = t < tiles;
    int te = valid ? t : tiles - 1;
    v8f acc[4];
    tile_gemm2(te, lane, feats, bidx, a1s, smaxs, w2s, acc);
    // x2 = relu(bn2(y2)); D-layout (lane=channel) -> A-layout (lane=point)
    #pragma unroll
    for (int nT = 0; nT < 4; ++nT)
      #pragma unroll
      for (int r = 0; r < 8; ++r) {
        float x = acc[nT][r] * s2[nT] + t2[nT];
        x = x > 0.f ? x : 0.f;
        x2s[wv][r + 8*hi][nT*16 + pl] = (_Float16)x;
      }
    __syncthreads();
    v8f o;
    #pragma unroll
    for (int j = 0; j < 8; ++j) o[j] = 0.f;
    #pragma unroll
    for (int kT = 0; kT < 2; ++kT) {
      FragA a;
      #pragma unroll
      for (int v = 0; v < 8; ++v) {
        int k = kT*32 + 2*(v & 3) + ((v >= 4) ? 16 : 0) + (hi ? 8 : 0);
        a.e[2*v]   = x2s[wv][pl][k];
        a.e[2*v+1] = x2s[wv][pl][k + 1];
      }
      FragB bf;
      const uint4* src = (const uint4*)(w3s + (kT*32 + lane)*8);
      bf.q[0] = src[0]; bf.q[1] = src[1];
      o = __builtin_amdgcn_wmma_f32_16x16x32_f16(
          false, a.v, false, bf.v, (short)0, o, false, false);
    }
    if (valid) {
      int pbase = te * 16;
      #pragma unroll
      for (int r = 0; r < 8; ++r)
        out[(pbase + r + 8*hi)*16 + pl] = o[r] + bias;
    }
    __syncthreads();
  }
}

// ---------------------------------------------------------------------------
extern "C" void kernel_launch(void* const* d_in, const int* in_sizes, int n_in,
                              void* d_out, int out_size, void* d_ws, size_t ws_size,
                              hipStream_t stream) {
  const float* feats = (const float*)d_in[0];
  const int*   bidx  = (const int*)  d_in[1];
  const float* W1    = (const float*)d_in[2];
  const float* g1    = (const float*)d_in[3];
  const float* b1    = (const float*)d_in[4];
  const float* W2    = (const float*)d_in[5];
  const float* g2    = (const float*)d_in[6];
  const float* b2    = (const float*)d_in[7];
  const float* W3    = (const float*)d_in[8];
  const float* b3    = (const float*)d_in[9];
  float* out = (float*)d_out;
  unsigned* wsU = (unsigned*)d_ws;
  float*    wsF = (float*)d_ws;
  int n = in_sizes[0] / 3;
  (void)n_in; (void)out_size; (void)ws_size;

  k_init  <<<1, 256, 0, stream>>>(wsU);
  k_fstats<<<256, 256, 0, stream>>>(feats, wsF, n);
  k_prep  <<<1, 256, 0, stream>>>(W1, g1, b1, W2, W3, wsF, wsU, n);
  k_segmax<<<(n + 255) / 256, 256, 0, stream>>>(feats, bidx, wsF, wsU, n);
  k_y2stats<<<512, 128, 0, stream>>>(feats, bidx, wsU, wsF, n);
  k_final <<<512, 128, 0, stream>>>(feats, bidx, g2, b2, b3, wsU, wsF, out, n);
}
